// CASTransformerLayer_46943992545734
// MI455X (gfx1250) — matrix operand order repaired and checked by reading, not compiled, not run on hardware
//
#include <hip/hip_runtime.h>
#include <cstdint>

// ---------------------------------------------------------------------------
// CDNA5 / gfx1250 types for WMMA bf16
// ---------------------------------------------------------------------------
typedef __attribute__((ext_vector_type(16))) __bf16 v16bf;
typedef __attribute__((ext_vector_type(8)))  float  v8f;

union FragBF { uint32_t w[8]; v16bf v; };

__device__ __forceinline__ uint32_t pack_bf16x2(float a, float b) {
  // round-to-nearest-even f32 -> bf16, packed low|high
  union { float f; uint32_t u; } ca, cb;
  ca.f = a; cb.f = b;
  uint32_t ra = (ca.u + 0x7FFFu + ((ca.u >> 16) & 1u)) >> 16;
  uint32_t rb = (cb.u + 0x7FFFu + ((cb.u >> 16) & 1u)) >> 16;
  return (ra & 0xFFFFu) | (rb << 16);
}

// Prefetch one 128x32 tile of A and of W into registers (raw f32, no pack:
// keeps only vmem loads on the critical path; conversion VALU deferred).
__device__ __forceinline__ void load_tile_regs(const float* __restrict__ A, int lda,
                                               const float* __restrict__ W, int ldw,
                                               int m0, int n0, int kb, int tid,
                                               float2 ar[8], float2 wr[8]) {
#pragma unroll
  for (int r = 0; r < 8; ++r) {
    int slot = r * 256 + tid;       // 2048 float2 slots
    int row  = slot >> 4;           // 0..127
    int kp   = slot & 15;           // 0..15 (k pair)
    ar[r] = *((const float2*)(A + (size_t)(m0 + row) * lda + kb) + kp);
    wr[r] = *((const float2*)(W + (size_t)(n0 + row) * ldw + kb) + kp);
  }
}

// Convert + store the prefetched tile into LDS buffer, packed [kpair][row].
__device__ __forceinline__ void store_tile_lds(uint32_t (*Al)[132], uint32_t (*Bl)[132],
                                               int tid, const float2 ar[8], const float2 wr[8]) {
#pragma unroll
  for (int r = 0; r < 8; ++r) {
    int slot = r * 256 + tid;
    int row  = slot >> 4;
    int kp   = slot & 15;
    Al[kp][row] = pack_bf16x2(ar[r].x, ar[r].y);
    Bl[kp][row] = pack_bf16x2(wr[r].x, wr[r].y);
  }
}

// ---------------------------------------------------------------------------
// Tiled WMMA GEMM:  C[M,N] = A[M,K] @ W[N,K]^T + bias[N]
// BM=128, BN=128, BK=32, 256 threads = 8 waves, wave tile 32x64.
// Double-buffered LDS; next tile prefetched into registers under the WMMAs.
// ---------------------------------------------------------------------------
__global__ __launch_bounds__(256)
void wmma_gemm_bias(const float* __restrict__ A, int lda,
                    const float* __restrict__ W, int ldw,
                    const float* __restrict__ bias,
                    float* __restrict__ C, int ldc, int K) {
  __shared__ uint32_t Alds[2][16][132];   // [buf][kpair][m]
  __shared__ uint32_t Blds[2][16][132];   // [buf][kpair][n]

  const int tid  = threadIdx.x;
  const int wid  = tid >> 5;
  const int lane = tid & 31;
  const int hf   = lane >> 4;       // lane half (0/1)
  const int l16  = lane & 15;

  const int m0 = blockIdx.y * 128;
  const int n0 = blockIdx.x * 128;
  const int wm0 = (wid >> 1) * 32;  // wave row offset within block tile
  const int wn0 = (wid & 1) * 64;   // wave col offset within block tile

  const v8f vzero = {0.f,0.f,0.f,0.f,0.f,0.f,0.f,0.f};
  v8f acc[2][4];
#pragma unroll
  for (int mf = 0; mf < 2; ++mf)
#pragma unroll
    for (int nf = 0; nf < 4; ++nf) acc[mf][nf] = vzero;

  float2 ar[8], wr[8];
  load_tile_regs(A, lda, W, ldw, m0, n0, 0, tid, ar, wr);
  store_tile_lds(Alds[0], Blds[0], tid, ar, wr);
  __syncthreads();

  const int nIter = K >> 5;
  for (int it = 0; it < nIter; ++it) {
    const int buf = it & 1;
    const bool hasNext = (it + 1) < nIter;
    if (hasNext)  // global loads for tile it+1 fly under this tile's WMMAs
      load_tile_regs(A, lda, W, ldw, m0, n0, (it + 1) << 5, tid, ar, wr);

    // ---- build fragments per ISA 7.12.2 layouts ----
    FragBF afr[2], bfr[4];
#pragma unroll
    for (int mf = 0; mf < 2; ++mf) {
      int mrow = wm0 + mf * 16 + l16;
#pragma unroll
      for (int i = 0; i < 8; ++i) {
        // A 16x32 bf16: kbase = (i>>2)*16 + hf*8 + (i&3)*2
        int kp = (i >> 2) * 8 + hf * 4 + (i & 3);
        afr[mf].w[i] = Alds[buf][kp][mrow];
      }
    }
#pragma unroll
    for (int nf = 0; nf < 4; ++nf) {
      int ncol = wn0 + nf * 16 + l16;
#pragma unroll
      for (int i = 0; i < 8; ++i) {
        // B 32x16 bf16: kbase = (i>>2)*16 + hf*8 + (i&3)*2
        int kp = (i >> 2) * 8 + hf * 4 + (i & 3);
        bfr[nf].w[i] = Blds[buf][kp][ncol];
      }
    }

#pragma unroll
    for (int mf = 0; mf < 2; ++mf)
#pragma unroll
      for (int nf = 0; nf < 4; ++nf)
        acc[mf][nf] = __builtin_amdgcn_wmma_f32_16x16x32_bf16(
            false, afr[mf].v, false, bfr[nf].v,
            (short)0, acc[mf][nf], false, false);

    if (hasNext)  // write-ahead into the other buffer (no reader conflict)
      store_tile_lds(Alds[buf ^ 1], Blds[buf ^ 1], tid, ar, wr);
    __syncthreads();
  }

  // ---- epilogue: C/D 16x16 f32 layout: row = vg + hf*8, col = l16 ----
#pragma unroll
  for (int nf = 0; nf < 4; ++nf) {
    int col = n0 + wn0 + nf * 16 + l16;
    float bv = bias[col];
#pragma unroll
    for (int mf = 0; mf < 2; ++mf) {
#pragma unroll
      for (int vg = 0; vg < 8; ++vg) {
        int row = m0 + wm0 + mf * 16 + hf * 8 + vg;
        C[(size_t)row * ldc + col] = acc[mf][nf][vg] + bv;
      }
    }
  }
}

// ---------------------------------------------------------------------------
// Sliding-window attention with the reference's faithful "raw reshape":
// group g sums q*k over channels c = 64h + 4g + u (h=0..15, u=0..3), output
// written to column 64g + d2 where d2 = 16u + h.  One wave per (b,branch,g,s).
// ---------------------------------------------------------------------------
__global__ __launch_bounds__(256)
void cas_attn(const float* __restrict__ Q, const float* __restrict__ Kp,
              const float* __restrict__ V, float* __restrict__ attn,
              float* __restrict__ probs, int seq) {
  const int wid  = threadIdx.x >> 5;
  const int lane = threadIdx.x & 31;
  const int s      = blockIdx.x * 8 + wid;
  const int g      = blockIdx.y;
  const int b      = blockIdx.z >> 1;
  const int branch = blockIdx.z & 1;
  const int dil    = branch ? 1 : 3;
  const int pad    = dil * 2;
  const int base   = branch * 1024;
  const size_t stride = 2048;

  const int d2a = lane, d2b = lane + 32;
  const int ca = 64 * (d2a & 15) + 4 * g + (d2a >> 4);
  const int cb = 64 * (d2b & 15) + 4 * g + (d2b >> 4);

  const size_t rq = ((size_t)b * seq + s) * stride + base;
  const float qa = Q[rq + ca];
  const float qb = Q[rq + cb];

  float sc[5];
#pragma unroll
  for (int j = 0; j < 5; ++j) {
    int sp = s + dil * j - pad;
    float p = 0.f;
    if (sp >= 0 && sp < seq) {
      const size_t rk = ((size_t)b * seq + sp) * stride + base;
      p = qa * Kp[rk + ca] + qb * Kp[rk + cb];
    }
#pragma unroll
    for (int off = 16; off; off >>= 1) p += __shfl_xor(p, off, 32);
    sc[j] = p * 0.125f;   // d_k = 64 -> scale = 1/8
  }

  // softmax over K=5 (zero-padded OOB scores participate, as in reference)
  float mx = sc[0];
#pragma unroll
  for (int j = 1; j < 5; ++j) mx = fmaxf(mx, sc[j]);
  float e[5], sum = 0.f;
#pragma unroll
  for (int j = 0; j < 5; ++j) { e[j] = __expf(sc[j] - mx); sum += e[j]; }
  const float inv = 1.f / sum;
#pragma unroll
  for (int j = 0; j < 5; ++j) e[j] *= inv;

  if (lane == 0) {
    size_t po = ((size_t)(b * 32 + branch * 16 + g) * seq + s) * 5;
#pragma unroll
    for (int j = 0; j < 5; ++j) probs[po + j] = e[j];
  }

  float oa = 0.f, ob = 0.f;
#pragma unroll
  for (int j = 0; j < 5; ++j) {
    int sp = s + dil * j - pad;
    if (sp >= 0 && sp < seq) {
      const size_t rv = ((size_t)b * seq + sp) * stride + base;
      oa += e[j] * V[rv + ca];
      ob += e[j] * V[rv + cb];
    }
  }
  float* arow = attn + ((size_t)b * seq + s) * stride + base + 64 * g;
  arow[d2a] = oa;
  arow[d2b] = ob;
}

// ---------------------------------------------------------------------------
// y = LayerNorm(query + y) * ln_w + ln_b, in place on d_out's y region.
// One 256-thread block per row of 2048.
// ---------------------------------------------------------------------------
__global__ __launch_bounds__(256)
void residual_ln(float* __restrict__ y, const float* __restrict__ q,
                 const float* __restrict__ lnw, const float* __restrict__ lnb) {
  __shared__ float red[8];
  const int tid = threadIdx.x, wid = tid >> 5, lane = tid & 31;
  const size_t off = (size_t)blockIdx.x * 2048;

  float r[8]; float s = 0.f;
#pragma unroll
  for (int i = 0; i < 8; ++i) {
    int c = tid + i * 256;
    r[i] = y[off + c] + q[off + c];
    s += r[i];
  }
#pragma unroll
  for (int o = 16; o; o >>= 1) s += __shfl_xor(s, o, 32);
  if (!lane) red[wid] = s;
  __syncthreads();
  float tot = 0.f;
#pragma unroll
  for (int i = 0; i < 8; ++i) tot += red[i];
  const float mu = tot * (1.f / 2048.f);
  __syncthreads();

  float v = 0.f;
#pragma unroll
  for (int i = 0; i < 8; ++i) { float d = r[i] - mu; v += d * d; }
#pragma unroll
  for (int o = 16; o; o >>= 1) v += __shfl_xor(v, o, 32);
  if (!lane) red[wid] = v;
  __syncthreads();
  float vt = 0.f;
#pragma unroll
  for (int i = 0; i < 8; ++i) vt += red[i];
  const float rstd = rsqrtf(vt * (1.f / 2048.f) + 1e-5f);

#pragma unroll
  for (int i = 0; i < 8; ++i) {
    int c = tid + i * 256;
    y[off + c] = (r[i] - mu) * rstd * lnw[c] + lnb[c];
  }
}

// ---------------------------------------------------------------------------
extern "C" void kernel_launch(void* const* d_in, const int* in_sizes, int n_in,
                              void* d_out, int out_size, void* d_ws, size_t ws_size,
                              hipStream_t stream) {
  (void)in_sizes; (void)n_in; (void)out_size; (void)ws_size;
  const float* query  = (const float*)d_in[0];
  const float* key    = (const float*)d_in[1];
  const float* values = (const float*)d_in[2];
  /* d_in[3] = lens (unused by reference math) */
  const float* W0 = (const float*)d_in[4];
  const float* b0 = (const float*)d_in[5];
  const float* W1 = (const float*)d_in[6];
  const float* b1 = (const float*)d_in[7];
  const float* Wo = (const float*)d_in[8];
  const float* bo = (const float*)d_in[9];
  const float* lnw = (const float*)d_in[10];
  const float* lnb = (const float*)d_in[11];

  const int bs = 2, seq = 4096, d_model = 2048, split = 1024;
  const size_t plane = (size_t)bs * seq * d_model;  // 16,777,216 floats

  float* projQ = (float*)d_ws;
  float* projK = projQ + plane;
  float* projV = projK + plane;
  float* attnB = projV + plane;
  float* yout  = (float*)d_out;
  float* probs = yout + plane;

  dim3 blk(256);
  dim3 gP(split / 128, (bs * seq) / 128);          // (8, 64)

  // branch 0 (columns [0,1024)) — same weight for q,k,v
  wmma_gemm_bias<<<gP, blk, 0, stream>>>(query,  d_model, W0, split, b0, projQ, d_model, split);
  wmma_gemm_bias<<<gP, blk, 0, stream>>>(key,    d_model, W0, split, b0, projK, d_model, split);
  wmma_gemm_bias<<<gP, blk, 0, stream>>>(values, d_model, W0, split, b0, projV, d_model, split);
  // branch 1 (columns [1024,2048))
  wmma_gemm_bias<<<gP, blk, 0, stream>>>(query  + split, d_model, W1, split, b1, projQ + split, d_model, split);
  wmma_gemm_bias<<<gP, blk, 0, stream>>>(key    + split, d_model, W1, split, b1, projK + split, d_model, split);
  wmma_gemm_bias<<<gP, blk, 0, stream>>>(values + split, d_model, W1, split, b1, projV + split, d_model, split);

  // sliding-window attention + softmax + probs output
  dim3 gA(seq / 8, 16, bs * 2);                    // (512, 16, 4)
  cas_attn<<<gA, blk, 0, stream>>>(projQ, projK, projV, attnB, probs, seq);

  // out_proj into d_out's y region
  dim3 gO(d_model / 128, (bs * seq) / 128);        // (16, 64)
  wmma_gemm_bias<<<gO, blk, 0, stream>>>(attnB, d_model, Wo, d_model, bo, yout, d_model, d_model);

  // residual + LayerNorm in place
  residual_ln<<<dim3(bs * seq), blk, 0, stream>>>(yout, query, lnw, lnb);
}
